// GAT_59287728554427
// MI455X (gfx1250) — matrix operand order repaired and checked by reading, not compiled, not run on hardware
//
#include <hip/hip_runtime.h>
#include <hip/hip_bf16.h>
#include <math.h>

typedef __bf16 bf16_t;
typedef __attribute__((ext_vector_type(16))) __bf16 v16bf;
typedef __attribute__((ext_vector_type(8)))  float  v8f;

#define NEG_SLOPE 0.2f
#define BN_EPS 1e-5f

// ---- order-preserving float<->uint encoding so float max maps to atomic_max_u32 ----
__device__ __forceinline__ unsigned fenc(float f) {
  unsigned u = __float_as_uint(f);
  return (u & 0x80000000u) ? ~u : (u | 0x80000000u);
}
__device__ __forceinline__ float fdec(unsigned u) {
  unsigned v = (u & 0x80000000u) ? (u ^ 0x80000000u) : ~u;
  return __uint_as_float(v);
}

// =================== BatchNorm ===================
__global__ void k_bn_sum(const float* __restrict__ x, float* __restrict__ sum,
                         float* __restrict__ sq, int N) {
  int c = threadIdx.x;  // 0..127
  float s = 0.f, q = 0.f;
  for (int r = blockIdx.x; r < N; r += gridDim.x) {
    float v = x[(size_t)r * 128 + c];
    s += v; q += v * v;
  }
  atomicAdd(&sum[c], s);
  atomicAdd(&sq[c], q);
}

__global__ void k_bn_stats(const float* __restrict__ sum, const float* __restrict__ sq,
                           const float* __restrict__ gamma, const float* __restrict__ beta,
                           float* __restrict__ scale, float* __restrict__ shift, int N) {
  int c = threadIdx.x;
  float mu  = sum[c] / (float)N;
  float var = sq[c] / (float)N - mu * mu;
  float sc  = gamma[c] * rsqrtf(var + BN_EPS);
  scale[c] = sc;
  shift[c] = beta[c] - mu * sc;
}

__global__ void k_bn_apply(const float* __restrict__ x, const float* __restrict__ scale,
                           const float* __restrict__ shift, bf16_t* __restrict__ act,
                           size_t total) {
  size_t idx = (size_t)blockIdx.x * blockDim.x + threadIdx.x;
  size_t stride = (size_t)gridDim.x * blockDim.x;
  for (; idx < total; idx += stride) {
    int c = (int)(idx & 127);
    act[idx] = (bf16_t)(x[idx] * scale[c] + shift[c]);
  }
}

// ====== weight f32[K][F] -> bf16 transposed BT[F][K] (fragment-friendly order) ======
__global__ void k_w_to_bt(const float* __restrict__ in, bf16_t* __restrict__ out,
                          int K, int F) {
  int idx = blockIdx.x * blockDim.x + threadIdx.x;
  int total = K * F;
  for (; idx < total; idx += gridDim.x * blockDim.x) {
    int k = idx / F, c = idx - k * F;
    out[(size_t)c * K + k] = (bf16_t)in[idx];
  }
}

// =================== GEMM: C[M,Nout] = A[M,K](bf16) x W (as BT[Nout][K] bf16), f32 out ===
// block = 128 threads = 4 waves; block tile = 16 rows x 64 cols; wave tile = 16x16 WMMA.
// A strip (16 x K) staged once in LDS (shared by all 4 waves); B fragments read directly
// from the transposed, L1/L2-resident BT as contiguous b128 chunks.
__global__ void __launch_bounds__(128)
k_gemm_bf16_wmma(const bf16_t* __restrict__ A, const bf16_t* __restrict__ BT,
                 float* __restrict__ C, int M, int K, int Nout) {
  __shared__ bf16_t As[16][264];  // K <= 256, row = 528B (16B-aligned chunks)

  const int tid  = threadIdx.x;
  const int wave = tid >> 5;
  const int lane = tid & 31;
  const int row0 = blockIdx.x * 16;
  const int col0 = blockIdx.y * 64;
  const int n0   = wave * 16;

  // stage the 16 x K A-strip once: each thread K/64 uint4 chunks
  const int nchunk = K >> 6;  // K=128 -> 2, K=256 -> 4
  for (int i = 0; i < nchunk; ++i) {
    int l  = (tid + i * 128) * 8;
    int m  = l / K;
    int kk = l - m * K;
    int r  = row0 + m; if (r >= M) r = M - 1;
    *(uint4*)&As[m][kk] = *(const uint4*)(A + (size_t)r * K + kk);
  }
  __syncthreads();

  const int m  = lane & 15;
  const int hi = lane >> 4;
  const int n  = lane & 15;
  const bf16_t* btcol = BT + (size_t)(col0 + n0 + n) * K;

  v8f acc = {};
  for (int k0 = 0; k0 < K; k0 += 32) {
    if (k0 + 32 < K) __builtin_prefetch(btcol + k0 + 32 + hi * 16, 0, 1);
    union { v16bf v; uint4 u[2]; } a, b;
    // A fragment: lane = row m; VGPR pairs hold K = (hi*8 + 2i, +1) and K = 16 + hi*8 + 2i
    a.u[0] = *(const uint4*)&As[m][k0 + hi * 8];
    a.u[1] = *(const uint4*)&As[m][k0 + 16 + hi * 8];
    // B fragment: lane = col n; VGPR pairs hold K = (hi*8 + 2i, +1) and K = 16 + hi*8 + 2i
    b.u[0] = *(const uint4*)(btcol + k0 + hi * 8);
    b.u[1] = *(const uint4*)(btcol + k0 + 16 + hi * 8);
    acc = __builtin_amdgcn_wmma_f32_16x16x32_bf16(false, a.v, false, b.v, (short)0, acc,
                                                  false, false);
  }

  // store C: lane holds column n; VGPR r -> row r + (hi ? 8 : 0)
#pragma unroll
  for (int r = 0; r < 8; ++r) {
    int mm = r + (hi ? 8 : 0);
    if (row0 + mm < M) C[(size_t)(row0 + mm) * Nout + col0 + n0 + n] = acc[r];
  }
}

// =================== attention el/er ===================
__global__ void k_eler(const float* __restrict__ feat, const float* __restrict__ al,
                       const float* __restrict__ ar, float* __restrict__ el,
                       float* __restrict__ er, int Nn, int H, int D) {
  int idx = blockIdx.x * blockDim.x + threadIdx.x;
  int total = Nn * H;
  for (; idx < total; idx += gridDim.x * blockDim.x) {
    int n = idx / H, h = idx - n * H;
    const float4* fp = (const float4*)(feat + (size_t)n * H * D + h * D);
    const float4* ap = (const float4*)(al + h * D);
    const float4* bp = (const float4*)(ar + h * D);
    float sl = 0.f, sr = 0.f;
#pragma unroll 4
    for (int i = 0; i < 16; ++i) {  // D==64 -> 16 float4
      float4 f = fp[i], A = ap[i], Bv = bp[i];
      sl += f.x * A.x + f.y * A.y + f.z * A.z + f.w * A.w;
      sr += f.x * Bv.x + f.y * Bv.y + f.z * Bv.z + f.w * Bv.w;
    }
    el[idx] = sl;
    er[idx] = sr;
  }
}

// =================== edge softmax passes ===================
__global__ void k_edge_max(const int* __restrict__ src, const int* __restrict__ dst,
                           const float* __restrict__ el, const float* __restrict__ er,
                           unsigned* __restrict__ nmax, int E, int H) {
  int idx = blockIdx.x * blockDim.x + threadIdx.x;
  int total = E * H;
  for (; idx < total; idx += gridDim.x * blockDim.x) {
    int e = idx / H, h = idx - e * H;
    int s = src[e], d = dst[e];
    float v = el[s * H + h] + er[d * H + h];
    v = v > 0.f ? v : NEG_SLOPE * v;
    atomicMax(&nmax[d * H + h], fenc(v));
  }
}

__global__ void k_edge_expsum(const int* __restrict__ src, const int* __restrict__ dst,
                              const float* __restrict__ el, const float* __restrict__ er,
                              const unsigned* __restrict__ nmax, float* __restrict__ nsum,
                              int E, int H) {
  int idx = blockIdx.x * blockDim.x + threadIdx.x;
  int total = E * H;
  for (; idx < total; idx += gridDim.x * blockDim.x) {
    int e = idx / H, h = idx - e * H;
    int s = src[e], d = dst[e];
    float v = el[s * H + h] + er[d * H + h];
    v = v > 0.f ? v : NEG_SLOPE * v;
    float ex = __expf(v - fdec(nmax[d * H + h]));
    atomicAdd(&nsum[d * H + h], ex);
  }
}

// one block-slice per edge; thread per channel c in [0,F)
__global__ void k_scatter(const int* __restrict__ src, const int* __restrict__ dst,
                          const float* __restrict__ el, const float* __restrict__ er,
                          const unsigned* __restrict__ nmax, const float* __restrict__ nsum,
                          const float* __restrict__ feat, float* __restrict__ out,
                          int E, int H, int D, int F) {
  int c = threadIdx.x;
  int h = c / D;
  for (int e = blockIdx.x; e < E; e += gridDim.x) {
    int s = src[e], d = dst[e];
    float v = el[s * H + h] + er[d * H + h];
    v = v > 0.f ? v : NEG_SLOPE * v;
    float ex = __expf(v - fdec(nmax[d * H + h]));
    float a  = ex / (nsum[d * H + h] + 1e-9f);
    atomicAdd(&out[(size_t)d * F + c], a * feat[(size_t)s * F + c]);
  }
}

// =================== finalize ===================
__global__ void k_bias_relu_bf(const float* __restrict__ outacc, const float* __restrict__ bias,
                               bf16_t* __restrict__ act, size_t total, int F) {
  size_t idx = (size_t)blockIdx.x * blockDim.x + threadIdx.x;
  size_t stride = (size_t)gridDim.x * blockDim.x;
  for (; idx < total; idx += stride) {
    int c = (int)(idx % (size_t)F);
    float v = outacc[idx] + bias[c];
    v = v > 0.f ? v : 0.f;
    act[idx] = (bf16_t)v;
  }
}

__global__ void k_bias_out(const float* __restrict__ outacc, const float* __restrict__ bias,
                           float* __restrict__ dout, size_t total, int F) {
  size_t idx = (size_t)blockIdx.x * blockDim.x + threadIdx.x;
  size_t stride = (size_t)gridDim.x * blockDim.x;
  for (; idx < total; idx += stride) {
    int c = (int)(idx % (size_t)F);
    dout[idx] = outacc[idx] + bias[c];
  }
}

// =================== host ===================
extern "C" void kernel_launch(void* const* d_in, const int* in_sizes, int n_in,
                              void* d_out, int out_size, void* d_ws, size_t ws_size,
                              hipStream_t stream) {
  const float* x        = (const float*)d_in[0];
  const int*   src      = (const int*)d_in[1];
  const int*   dst      = (const int*)d_in[2];
  const float* bn_gamma = (const float*)d_in[3];
  const float* bn_beta  = (const float*)d_in[4];
  const float* W[4]  = {(const float*)d_in[5],  (const float*)d_in[9],
                        (const float*)d_in[13], (const float*)d_in[17]};
  const float* al[4] = {(const float*)d_in[6],  (const float*)d_in[10],
                        (const float*)d_in[14], (const float*)d_in[18]};
  const float* ar[4] = {(const float*)d_in[7],  (const float*)d_in[11],
                        (const float*)d_in[15], (const float*)d_in[19]};
  const float* bi[4] = {(const float*)d_in[8],  (const float*)d_in[12],
                        (const float*)d_in[16], (const float*)d_in[20]};

  const int N = in_sizes[0] / 128;
  const int E = in_sizes[1];

  // workspace bump allocator
  size_t off = 0;
  char* ws = (char*)d_ws;
  auto alloc = [&](size_t bytes) -> void* {
    void* p = ws + off;
    off += (bytes + 255) & ~(size_t)255;
    return p;
  };

  bf16_t* actbf  = (bf16_t*)alloc((size_t)N * 256 * sizeof(bf16_t));
  float*  feat   = (float*) alloc((size_t)N * 256 * sizeof(float));
  float*  outacc = (float*) alloc((size_t)N * 256 * sizeof(float));
  float*  el     = (float*) alloc((size_t)N * 4 * sizeof(float));
  float*  er     = (float*) alloc((size_t)N * 4 * sizeof(float));
  unsigned* nmax = (unsigned*)alloc((size_t)N * 4 * sizeof(unsigned));
  float*  nsum   = (float*) alloc((size_t)N * 4 * sizeof(float));
  bf16_t* wbt[4];
  const int wK[4] = {128, 256, 256, 256};
  const int wF[4] = {256, 256, 256, 64};
  for (int i = 0; i < 4; ++i)
    wbt[i] = (bf16_t*)alloc((size_t)wK[i] * wF[i] * sizeof(bf16_t));
  float* bnacc = (float*)alloc(256 * sizeof(float));  // [0:128)=sum, [128:256)=sumsq
  float* bnss  = (float*)alloc(256 * sizeof(float));  // [0:128)=scale, [128:256)=shift

  // ---- BatchNorm ----
  hipMemsetAsync(bnacc, 0, 256 * sizeof(float), stream);
  k_bn_sum<<<512, 128, 0, stream>>>(x, bnacc, bnacc + 128, N);
  k_bn_stats<<<1, 128, 0, stream>>>(bnacc, bnacc + 128, bn_gamma, bn_beta, bnss, bnss + 128, N);
  {
    size_t total = (size_t)N * 128;
    k_bn_apply<<<(int)((total + 255) / 256), 256, 0, stream>>>(x, bnss, bnss + 128, actbf, total);
  }

  // ---- weight convert + transpose ----
  for (int i = 0; i < 4; ++i) {
    int tot = wK[i] * wF[i];
    k_w_to_bt<<<(tot + 255) / 256, 256, 0, stream>>>(W[i], wbt[i], wK[i], wF[i]);
  }

  // ---- layers ----
  auto run_layer = [&](int li, int K, int F, int H, int D, bool last) {
    dim3 g((N + 15) / 16, F / 64);
    k_gemm_bf16_wmma<<<g, 128, 0, stream>>>(actbf, wbt[li], feat, N, K, F);

    int NH = N * H;
    k_eler<<<(NH + 255) / 256, 256, 0, stream>>>(feat, al[li], ar[li], el, er, N, H, D);

    hipMemsetAsync(outacc, 0, (size_t)N * 256 * sizeof(float), stream);
    hipMemsetAsync(nmax, 0, (size_t)N * 4 * sizeof(unsigned), stream);  // enc(-NaN): below all reals
    hipMemsetAsync(nsum, 0, (size_t)N * 4 * sizeof(float), stream);

    int totEH = E * H;
    int gb = (totEH + 255) / 256;
    k_edge_max<<<gb, 256, 0, stream>>>(src, dst, el, er, nmax, E, H);
    k_edge_expsum<<<gb, 256, 0, stream>>>(src, dst, el, er, nmax, nsum, E, H);
    k_scatter<<<8192, F, 0, stream>>>(src, dst, el, er, nmax, nsum, feat, outacc, E, H, D, F);

    size_t total = (size_t)N * F;
    if (!last)
      k_bias_relu_bf<<<(int)((total + 255) / 256), 256, 0, stream>>>(outacc, bi[li], actbf, total, F);
    else
      k_bias_out<<<(int)((total + 255) / 256), 256, 0, stream>>>(outacc, bi[li], (float*)d_out, total, F);
  };

  run_layer(0, 128, 256, 4, 64, false);
  run_layer(1, 256, 256, 4, 64, false);
  run_layer(2, 256, 256, 4, 64, false);
  run_layer(3, 256, 64, 1, 64, true);

  (void)n_in; (void)out_size; (void)ws_size;
}
